// GravitonTransformerBlock_15719580303661
// MI455X (gfx1250) — matrix operand order repaired and checked by reading, not compiled, not run on hardware
//
#include <hip/hip_runtime.h>
#include <hip/hip_bf16.h>

// ---------------- problem constants (match reference) ----------------
#define B_SZ   2
#define S_SZ   2048
#define DM_SZ  2048
#define NH_SZ  16
#define NKV_SZ 4
#define HD_SZ  128
#define DFF_SZ 8192
#define KKEEP  (DFF_SZ / 2)
#define T_SZ   (B_SZ * S_SZ)          // 4096 tokens
#define QDIM   (NH_SZ * HD_SZ)        // 2048
#define KVDIM  (NKV_SZ * HD_SZ)       // 512

typedef __attribute__((ext_vector_type(16))) __bf16        v16bf;
typedef __attribute__((ext_vector_type(8)))  float         v8f;
typedef __attribute__((ext_vector_type(8)))  unsigned int  u32x8;

// ---------------- helpers ----------------
__device__ __forceinline__ unsigned short f32_to_bf16(float f) {
    unsigned u = __builtin_bit_cast(unsigned, f);
    unsigned r = u + 0x7FFFu + ((u >> 16) & 1u);   // round-to-nearest-even
    return (unsigned short)(r >> 16);
}
__device__ __forceinline__ float bf16_to_f32(unsigned short h) {
    unsigned u = ((unsigned)h) << 16;
    return __builtin_bit_cast(float, u);
}

// A-matrix fragment (16x32 bf16, ISA 7.12.2): rowp already points at this
// lane's row (m = lane&15). k(j) = k0 + 2*j2 + (j2>=4 ? 8 : 0) + 8*(lane>>4)
__device__ __forceinline__ v16bf load_afrag(const unsigned short* rowp, int k0, int lane) {
    u32x8 r;
    const int kh = 8 * (lane >> 4);
#pragma unroll
    for (int j2 = 0; j2 < 8; ++j2) {
        int k = k0 + kh + 2 * j2 + ((j2 >= 4) ? 8 : 0);
        r[j2] = *(const unsigned int*)(rowp + k);
    }
    return __builtin_bit_cast(v16bf, r);
}

// B-matrix fragment (32x16 bf16): rowp already points at this lane's
// n-row (n = lane&15) of a row-major [N,K] array. k(j) = k0 + 8*(lane>>4) + j + (j>=8 ? 8 : 0)
__device__ __forceinline__ v16bf load_bfrag(const unsigned short* rowp, int k0, int lane) {
    u32x8 r;
    const int kb = k0 + 8 * (lane >> 4);
#pragma unroll
    for (int j2 = 0; j2 < 8; ++j2)
        r[j2] = *(const unsigned int*)(rowp + kb + 2 * j2 + ((j2 >= 4) ? 8 : 0));
    return __builtin_bit_cast(v16bf, r);
}

// ---------------- fp32 -> bf16 conversion ----------------
__global__ void cvt_bf16_kernel(const float* __restrict__ in,
                                unsigned short* __restrict__ out, size_t n) {
    size_t i = (size_t)blockIdx.x * blockDim.x + threadIdx.x;
    size_t stride = (size_t)gridDim.x * blockDim.x;
    for (; i < n; i += stride) out[i] = f32_to_bf16(in[i]);
}

// ---------------- RMSNorm (fp32 in, bf16 out), one block per token ----------------
__global__ void rmsnorm_kernel(const float* __restrict__ x, const float* __restrict__ w,
                               unsigned short* __restrict__ out, int ncols) {
    __shared__ float red[256];
    const int row = blockIdx.x;
    const float* xr = x + (size_t)row * ncols;
    float ss = 0.f;
    for (int i = threadIdx.x; i < ncols; i += blockDim.x) { float v = xr[i]; ss += v * v; }
    red[threadIdx.x] = ss;
    __syncthreads();
    for (int s = 128; s > 0; s >>= 1) {
        if ((int)threadIdx.x < s) red[threadIdx.x] += red[threadIdx.x + s];
        __syncthreads();
    }
    const float inv = rsqrtf(red[0] / (float)ncols + 1e-6f);
    for (int i = threadIdx.x; i < ncols; i += blockDim.x)
        out[(size_t)row * ncols + i] = f32_to_bf16(xr[i] * inv * w[i]);
}

// ---------------- bf16 WMMA GEMM: C[M,N] = A[M,K] @ W[N,K]^T ----------------
// Register-blocked: each wave computes 2x4 = eight 16x16 tiles (32 rows x 64 cols),
// 8 waves/block -> 64x256 block tile. Fragments are loop-local (no manual double
// buffering -> no spills); each iteration = 12 b128-load clause + 8 WMMAs.
// act: 0 = none, 1 = SiLU. residual fp32 [M,N] optional. outF/outB/outBT optional;
// outBT stores bf16 transposed as [N,M] (used to produce V^T for attention).
__global__ void gemm_bf16_wmma(const unsigned short* __restrict__ A,
                               const unsigned short* __restrict__ W,
                               int M, int N, int K,
                               const float* __restrict__ residual,
                               float* __restrict__ outF,
                               unsigned short* __restrict__ outB,
                               unsigned short* __restrict__ outBT,
                               int act) {
    const int lane = threadIdx.x & 31;
    const int wave = threadIdx.x >> 5;                  // 0..7 : 2 (M) x 4 (N)
    const int m_base = blockIdx.y * 64 + (wave >> 2) * 32;
    const int n_base = blockIdx.x * 256 + (wave & 3) * 64;
    if (m_base >= M || n_base >= N) return;

    const unsigned short* Arow[2];
    const unsigned short* Wrow[4];
#pragma unroll
    for (int mt = 0; mt < 2; ++mt)
        Arow[mt] = A + (size_t)(m_base + mt * 16 + (lane & 15)) * K;
#pragma unroll
    for (int nt = 0; nt < 4; ++nt)
        Wrow[nt] = W + (size_t)(n_base + nt * 16 + (lane & 15)) * K;

    v8f acc[2][4];
#pragma unroll
    for (int mt = 0; mt < 2; ++mt)
#pragma unroll
        for (int nt = 0; nt < 4; ++nt)
            acc[mt][nt] = (v8f){0.f, 0.f, 0.f, 0.f, 0.f, 0.f, 0.f, 0.f};

    for (int k0 = 0; k0 < K; k0 += 32) {
        v16bf af[2], bfr[4];
#pragma unroll
        for (int mt = 0; mt < 2; ++mt) af[mt] = load_afrag(Arow[mt], k0, lane);
#pragma unroll
        for (int nt = 0; nt < 4; ++nt) bfr[nt] = load_bfrag(Wrow[nt], k0, lane);
#pragma unroll
        for (int mt = 0; mt < 2; ++mt)
#pragma unroll
            for (int nt = 0; nt < 4; ++nt)
                acc[mt][nt] = __builtin_amdgcn_wmma_f32_16x16x32_bf16(
                    false, af[mt], false, bfr[nt], (short)0, acc[mt][nt], false, false);
    }

#pragma unroll
    for (int mt = 0; mt < 2; ++mt) {
#pragma unroll
        for (int nt = 0; nt < 4; ++nt) {
            const int n = n_base + nt * 16 + (lane & 15);
#pragma unroll
            for (int r = 0; r < 8; ++r) {
                const int m = m_base + mt * 16 + r + 8 * (lane >> 4);
                float v = acc[mt][nt][r];
                if (act == 1) v = v / (1.f + __expf(-v));                 // SiLU
                if (residual) v += residual[(size_t)m * N + n];
                if (outF)  outF[(size_t)m * N + n] = v;
                if (outB)  outB[(size_t)m * N + n] = f32_to_bf16(v);
                if (outBT) outBT[(size_t)n * M + m] = f32_to_bf16(v);
            }
        }
    }
}

// ---------------- GQA causal flash attention (bf16 WMMA), 1 wave per 16-q tile ------
// V is consumed pre-transposed in global memory: vT[kv_dim][T]. V^T tiles are staged
// into LDS with CDNA5 async global->LDS copies (global_load_async_to_lds_b128).
__global__ void attn_kernel(const unsigned short* __restrict__ qb,
                            const unsigned short* __restrict__ kb,
                            const unsigned short* __restrict__ vT,
                            unsigned short* __restrict__ ob) {
    __shared__ __align__(16) unsigned short ldsP[16 * 32];    // P tile [m][key]
    __shared__ __align__(16) unsigned short ldsVT[128 * 32];  // V^T tile [hd][key]

    const int lane = threadIdx.x;                 // blockDim.x == 32
    const int qt = blockIdx.x;                    // 0..S/16-1
    const int h  = blockIdx.y;                    // 0..NH-1
    const int bb = blockIdx.z;                    // 0..B-1
    const int kvh = h >> 2;                       // NH/NKV = 4
    const int q0 = qt * 16;
    const size_t tokq = (size_t)bb * S_SZ + q0;
    const float scale = 0.08838834764831845f;     // 1/sqrt(128)

    // LDS byte offset of ldsVT (low 32 bits of a generic LDS pointer = wave-relative offset)
    const unsigned ldsVT_base = (unsigned)(uintptr_t)(&ldsVT[0]);

    // preload Q fragments (16x128 -> 4 A-frags of 16x32)
    v16bf qf[4];
    {
        const unsigned short* Qrow = qb + (tokq + (lane & 15)) * QDIM + (size_t)h * HD_SZ;
#pragma unroll
        for (int kk = 0; kk < 4; ++kk) qf[kk] = load_afrag(Qrow, kk * 32, lane);
    }

    float mrow[8], lsum[8];
    v8f   oacc[8];
#pragma unroll
    for (int r = 0; r < 8; ++r) {
        mrow[r] = -3.0e38f; lsum[r] = 0.f;
        oacc[r] = (v8f){0.f, 0.f, 0.f, 0.f, 0.f, 0.f, 0.f, 0.f};
    }

    const int ntile = (q0 + 15) / 32 + 1;         // 32-key tiles to process
    for (int kt = 0; kt < ntile; ++kt) {
        const int key0 = kt * 32;

        // --- async-stage V^T tile: 128 hd-rows x 32 keys (8 KB), 16B per op/lane ---
        // vT row (kvh*128+hd) is contiguous over tokens -> contiguous b128 chunks.
#pragma unroll
        for (int it = 0; it < 16; ++it) {
            const int i  = it * 32 + lane;        // 0..511
            const int hd = i >> 2;                // 0..127
            const int kq = i & 3;                 // 8-key quad within the tile
            const unsigned lds_off = ldsVT_base + (unsigned)((hd * 32 + kq * 8) * 2);
            const unsigned long long ga = (unsigned long long)(uintptr_t)(
                vT + ((size_t)(kvh * HD_SZ + hd)) * T_SZ + (size_t)bb * S_SZ + key0 + kq * 8);
            asm volatile("global_load_async_to_lds_b128 %0, %1, off"
                         :: "v"(lds_off), "v"(ga) : "memory");
        }
        asm volatile("s_wait_asynccnt 0x0" ::: "memory");
        __syncthreads();

        // --- S = Q @ K^T (two 16x16 score frags; K read straight from global) ---
        v8f sf[2];
#pragma unroll
        for (int nt = 0; nt < 2; ++nt) {
            const int key = key0 + nt * 16 + (lane & 15);
            const unsigned short* Krow =
                kb + ((size_t)bb * S_SZ + key) * KVDIM + (size_t)kvh * HD_SZ;
            v8f s = {0.f, 0.f, 0.f, 0.f, 0.f, 0.f, 0.f, 0.f};
#pragma unroll
            for (int kk = 0; kk < 4; ++kk) {
                v16bf bfr = load_bfrag(Krow, kk * 32, lane);
                s = __builtin_amdgcn_wmma_f32_16x16x32_bf16(false, qf[kk], false, bfr,
                                                            (short)0, s, false, false);
            }
            sf[nt] = s;
        }

        // --- scale + causal mask + online softmax (xor-shuffle row reductions) ---
#pragma unroll
        for (int r = 0; r < 8; ++r) {
            const int m  = r + 8 * (lane >> 4);
            const int qi = q0 + m;
#pragma unroll
            for (int nt = 0; nt < 2; ++nt) {
                const int kj = key0 + nt * 16 + (lane & 15);
                float val = sf[nt][r] * scale;
                if (kj > qi) val = -3.0e38f;
                sf[nt][r] = val;
            }
            float t = fmaxf(sf[0][r], sf[1][r]);
            for (int off = 1; off < 16; off <<= 1) t = fmaxf(t, __shfl_xor(t, off, 32));
            const float newm  = fmaxf(mrow[r], t);
            const float alpha = __expf(mrow[r] - newm);
            const float p0 = __expf(sf[0][r] - newm);
            const float p1 = __expf(sf[1][r] - newm);
            float ps = p0 + p1;
            for (int off = 1; off < 16; off <<= 1) ps += __shfl_xor(ps, off, 32);
            lsum[r] = lsum[r] * alpha + ps;
            mrow[r] = newm;
#pragma unroll
            for (int c = 0; c < 8; ++c) oacc[c][r] *= alpha;
            ldsP[m * 32 +      (lane & 15)] = f32_to_bf16(p0);
            ldsP[m * 32 + 16 + (lane & 15)] = f32_to_bf16(p1);
        }
        __syncthreads();

        // --- O += P @ V (P A-frag from LDS, V 32x16 B-frags from V^T LDS) ---
        v16bf pf = load_afrag(&ldsP[(lane & 15) * 32], 0, lane);
#pragma unroll
        for (int c = 0; c < 8; ++c) {
            v16bf vf = load_bfrag(&ldsVT[(c * 16 + (lane & 15)) * 32], 0, lane);
            oacc[c] = __builtin_amdgcn_wmma_f32_16x16x32_bf16(false, pf, false, vf,
                                                              (short)0, oacc[c], false, false);
        }
        __syncthreads();
    }

    // normalize and write O (bf16, token-major [T, NH*HD])
#pragma unroll
    for (int r = 0; r < 8; ++r) {
        const float inv = 1.f / lsum[r];
        const int m = r + 8 * (lane >> 4);
#pragma unroll
        for (int c = 0; c < 8; ++c)
            ob[(tokq + m) * QDIM + (size_t)h * HD_SZ + c * 16 + (lane & 15)] =
                f32_to_bf16(oacc[c][r] * inv);
    }
}

// ---------------- exact top-k threshold (radix select on |f32| bits) + mask + up-mul ----------------
__global__ void topk_mask_kernel(const float* __restrict__ gate,
                                 const unsigned short* __restrict__ up,
                                 unsigned short* __restrict__ gact) {
    __shared__ unsigned hist[256];
    __shared__ unsigned sh_prefix;
    __shared__ int sh_k;
    const int row = blockIdx.x;
    const float* g = gate + (size_t)row * DFF_SZ;

    if (threadIdx.x == 0) { sh_prefix = 0u; sh_k = KKEEP; }
    __syncthreads();

    for (int pass = 0; pass < 4; ++pass) {
        const int shift = 24 - 8 * pass;
        hist[threadIdx.x] = 0u;
        __syncthreads();
        const unsigned prefix = sh_prefix;
        for (int i = threadIdx.x; i < DFF_SZ; i += 256) {
            unsigned u = __builtin_bit_cast(unsigned, g[i]) & 0x7FFFFFFFu;
            bool match = (pass == 0) || ((u >> (shift + 8)) == (prefix >> (shift + 8)));
            if (match) atomicAdd(&hist[(u >> shift) & 255u], 1u);
        }
        __syncthreads();
        if (threadIdx.x == 0) {
            int kneed = sh_k;
            unsigned acc = 0;
            int b = 255;
            for (; b > 0; --b) { acc += hist[b]; if ((int)acc >= kneed) break; }
            if ((int)acc < kneed) acc += hist[0];
            sh_k = kneed - (int)(acc - hist[b]);
            sh_prefix = prefix | ((unsigned)b << shift);
        }
        __syncthreads();
    }

    const unsigned thr = sh_prefix;   // bits of the KKEEP-th largest |gate|
    const unsigned short* u = up + (size_t)row * DFF_SZ;
    unsigned short* o = gact + (size_t)row * DFF_SZ;
    for (int i = threadIdx.x; i < DFF_SZ; i += 256) {
        const float gv = g[i];
        const unsigned mag = __builtin_bit_cast(unsigned, gv) & 0x7FFFFFFFu;
        const float val = (mag >= thr) ? gv * bf16_to_f32(u[i]) : 0.f;
        o[i] = f32_to_bf16(val);
    }
}

// ---------------- host-side orchestration ----------------
static inline size_t align256(size_t x) { return (x + 255) & ~(size_t)255; }

extern "C" void kernel_launch(void* const* d_in, const int* in_sizes, int n_in,
                              void* d_out, int out_size, void* d_ws, size_t ws_size,
                              hipStream_t stream) {
    const float* hidden = (const float*)d_in[0];
    const float* wq  = (const float*)d_in[1];
    const float* wk  = (const float*)d_in[2];
    const float* wv  = (const float*)d_in[3];
    const float* wo  = (const float*)d_in[4];
    const float* ln1 = (const float*)d_in[5];
    const float* ln2 = (const float*)d_in[6];
    const float* wg  = (const float*)d_in[7];
    const float* wu  = (const float*)d_in[8];
    const float* wd  = (const float*)d_in[9];
    float* out = (float*)d_out;                    // holds h, then h + ff

    char* ws = (char*)d_ws;
    size_t off = 0;
    auto carve = [&](size_t bytes) { char* p = ws + off; off += align256(bytes); return p; };
    unsigned short* wq_b  = (unsigned short*)carve((size_t)QDIM  * DM_SZ  * 2);
    unsigned short* wk_b  = (unsigned short*)carve((size_t)KVDIM * DM_SZ  * 2);
    unsigned short* wv_b  = (unsigned short*)carve((size_t)KVDIM * DM_SZ  * 2);
    unsigned short* wo_b  = (unsigned short*)carve((size_t)DM_SZ * QDIM   * 2);
    unsigned short* wg_b  = (unsigned short*)carve((size_t)DFF_SZ * DM_SZ * 2);
    unsigned short* wu_b  = (unsigned short*)carve((size_t)DFF_SZ * DM_SZ * 2);
    unsigned short* wd_b  = (unsigned short*)carve((size_t)DM_SZ * DFF_SZ * 2);
    unsigned short* xb    = (unsigned short*)carve((size_t)T_SZ * DM_SZ   * 2);
    unsigned short* qbuf  = (unsigned short*)carve((size_t)T_SZ * QDIM    * 2);
    unsigned short* kbuf  = (unsigned short*)carve((size_t)T_SZ * KVDIM   * 2);
    unsigned short* vTbuf = (unsigned short*)carve((size_t)KVDIM * T_SZ   * 2);  // transposed V
    unsigned short* obuf  = (unsigned short*)carve((size_t)T_SZ * QDIM    * 2);
    unsigned short* yb    = (unsigned short*)carve((size_t)T_SZ * DM_SZ   * 2);
    float*          gatef = (float*)         carve((size_t)T_SZ * DFF_SZ  * 4);
    unsigned short* upb   = (unsigned short*)carve((size_t)T_SZ * DFF_SZ  * 2);
    unsigned short* gactb = (unsigned short*)carve((size_t)T_SZ * DFF_SZ  * 2);
    (void)ws_size; (void)in_sizes; (void)n_in; (void)out_size;

    // 1) weights -> bf16
    cvt_bf16_kernel<<<2048, 256, 0, stream>>>(wq, wq_b, (size_t)QDIM * DM_SZ);
    cvt_bf16_kernel<<<2048, 256, 0, stream>>>(wk, wk_b, (size_t)KVDIM * DM_SZ);
    cvt_bf16_kernel<<<2048, 256, 0, stream>>>(wv, wv_b, (size_t)KVDIM * DM_SZ);
    cvt_bf16_kernel<<<2048, 256, 0, stream>>>(wo, wo_b, (size_t)DM_SZ * QDIM);
    cvt_bf16_kernel<<<4096, 256, 0, stream>>>(wg, wg_b, (size_t)DFF_SZ * DM_SZ);
    cvt_bf16_kernel<<<4096, 256, 0, stream>>>(wu, wu_b, (size_t)DFF_SZ * DM_SZ);
    cvt_bf16_kernel<<<4096, 256, 0, stream>>>(wd, wd_b, (size_t)DM_SZ * DFF_SZ);

    // 2) RMSNorm1 -> x (bf16)
    rmsnorm_kernel<<<T_SZ, 256, 0, stream>>>(hidden, ln1, xb, DM_SZ);

    // 3) Q/K/V projections (WMMA). V is written transposed for async attention staging.
    gemm_bf16_wmma<<<dim3(QDIM / 256,  T_SZ / 64), 256, 0, stream>>>(
        xb, wq_b, T_SZ, QDIM, DM_SZ, nullptr, nullptr, qbuf, nullptr, 0);
    gemm_bf16_wmma<<<dim3(KVDIM / 256, T_SZ / 64), 256, 0, stream>>>(
        xb, wk_b, T_SZ, KVDIM, DM_SZ, nullptr, nullptr, kbuf, nullptr, 0);
    gemm_bf16_wmma<<<dim3(KVDIM / 256, T_SZ / 64), 256, 0, stream>>>(
        xb, wv_b, T_SZ, KVDIM, DM_SZ, nullptr, nullptr, nullptr, vTbuf, 0);

    // 4) causal GQA flash attention (WMMA + async global->LDS V staging)
    attn_kernel<<<dim3(S_SZ / 16, NH_SZ, B_SZ), 32, 0, stream>>>(qbuf, kbuf, vTbuf, obuf);

    // 5) h = hidden + o @ wo^T  -> d_out (fp32)
    gemm_bf16_wmma<<<dim3(DM_SZ / 256, T_SZ / 64), 256, 0, stream>>>(
        obuf, wo_b, T_SZ, DM_SZ, QDIM, hidden, out, nullptr, nullptr, 0);

    // 6) RMSNorm2 -> y (bf16)
    rmsnorm_kernel<<<T_SZ, 256, 0, stream>>>(out, ln2, yb, DM_SZ);

    // 7) gate = silu(y @ wg^T) (fp32 for exact top-k), up = y @ wu^T (bf16)
    gemm_bf16_wmma<<<dim3(DFF_SZ / 256, T_SZ / 64), 256, 0, stream>>>(
        yb, wg_b, T_SZ, DFF_SZ, DM_SZ, nullptr, gatef, nullptr, nullptr, 1);
    gemm_bf16_wmma<<<dim3(DFF_SZ / 256, T_SZ / 64), 256, 0, stream>>>(
        yb, wu_b, T_SZ, DFF_SZ, DM_SZ, nullptr, nullptr, upb, nullptr, 0);

    // 8) exact top-k threshold + mask + elementwise product -> gact (bf16)
    topk_mask_kernel<<<T_SZ, 256, 0, stream>>>(gatef, upb, gactb);

    // 9) out = h + gact @ wd^T
    gemm_bf16_wmma<<<dim3(DM_SZ / 256, T_SZ / 64), 256, 0, stream>>>(
        gactb, wd_b, T_SZ, DM_SZ, DFF_SZ, out, out, nullptr, nullptr, 0);
}
